// Wav2Vec2BertFT_64201171140804
// MI455X (gfx1250) — matrix operand order repaired and checked by reading, not compiled, not run on hardware
//
#include <hip/hip_runtime.h>
#include <hip/hip_bf16.h>

// -------------------------------------------------------------------------
// Wav2Vec2Bert LSTM decoder, CDNA5 (gfx1250, wave32, WMMA bf16).
//
// Strategy (see analysis):
//  * Fold the [B,T,H]x[H,H] context GEMM through the T-reduction so hs is
//    read exactly once (64MB ~= 3us @ 23.3 TB/s HBM).
//  * 127 serial LSTM steps: [32,2048]x[2048,4096] bf16 WMMA GEMM per layer
//    (fused [inp|h] @ [w_ih^T ; w_hh^T]); weights converted+swizzled to the
//    WMMA-B fragment layout once per call (33.5MB bf16, L2-resident).
//  * Persistent cooperative kernel: 32 WGs x 256 threads (8 wave32).
//    Grid sync = one-shot atomic counters in ws (zeroed via hipMemsetAsync,
//    graph-capture safe). 3 grid barriers per step.
//  * Activation (A) buffers are double-buffered per step to avoid the
//    h_prev-read / h_new-write cross-WG race.
// -------------------------------------------------------------------------

typedef __attribute__((ext_vector_type(16))) __bf16   bf16x16;
typedef __attribute__((ext_vector_type(8)))  float    f32x8;
typedef __attribute__((ext_vector_type(4)))  unsigned u32x4;

union V16 { u32x4 q[2]; bf16x16 v; };

#define NWG 32
#define TPB 256

constexpr int B_ = 32, T_ = 512, H_ = 1024, NL_ = 2, NC_ = 61;
constexpr int K2 = 2 * H_;       // fused K = H (input) + H (hidden)
constexpr int NG = 4 * H_;       // 4096 gate columns
constexpr int NT = NG / 16;      // 256 N-tiles
constexpr int KT = K2 / 32;      // 64 K-tiles of 32

// workspace layout (bytes); total ~33.5 MiB
constexpr size_t BAR_OFF = 0;         // 4096 u32 one-shot barrier counters
constexpr size_t Y_OFF   = 16384;     // y       f32 [32][64]
constexpr size_t H1F_OFF = 24576;     // h1      f32 [32][1024]
constexpr size_t C_OFF   = 155648;    // c state f32 [2][32][1024]
constexpr size_t U_OFF   = 417792;    // u       f32 [2][2][32][1024]
constexpr size_t S_OFF   = 942080;    // 4 f32 (sum_t w_ch1 / w_cc1)
constexpr size_t A0_OFF  = 1048576;   // bf16 [2 bufs][32][2048]
constexpr size_t A1_OFF  = 1310720;   // bf16 [2 bufs][32][2048]
constexpr size_t WB_OFF  = 1572864;   // bf16 swizzled weights, 33554432 B

__device__ __forceinline__ void grid_sync(unsigned* bar, int idx) {
  __syncthreads();
  if (threadIdx.x == 0) {
    __hip_atomic_fetch_add(&bar[idx], 1u, __ATOMIC_ACQ_REL, __HIP_MEMORY_SCOPE_AGENT);
    while (__hip_atomic_load(&bar[idx], __ATOMIC_ACQUIRE, __HIP_MEMORY_SCOPE_AGENT)
           < (unsigned)NWG) {
      __builtin_amdgcn_s_sleep(2);
    }
  }
  __syncthreads();
}

__device__ __forceinline__ f32x8 wmma_bf16(bf16x16 a, bf16x16 b, f32x8 c) {
  return __builtin_amdgcn_wmma_f32_16x16x32_bf16(false, a, false, b, (short)0, c,
                                                 false, false);
}

// inp[b,:] = y[b,:] @ w_pre^T + b_pre  -> bf16 into A-buffer inp half
__device__ __forceinline__ void compute_inp(int b, const float* yg, const float* w_pre,
                                            const float* b_pre, __bf16* A0w) {
  for (int i = 0; i < 4; ++i) {
    int h = (int)threadIdx.x + i * TPB;
    float acc = b_pre[h];
    const float* wr = w_pre + (size_t)h * NC_;
    const float* yr = yg + b * 64;
    for (int c = 0; c < NC_; ++c) acc = fmaf(yr[c], wr[c], acc);
    A0w[b * K2 + h] = (__bf16)acc;
  }
}

// one LSTM layer: g = A[32,2048] x Wb[2048,4096] (bf16 WMMA), then cell update
__device__ __forceinline__ void lstm_layer(
    int l, const __bf16* Ag, const __bf16* Wb,
    const float* b_ih, const float* b_hh, float* cst,
    __bf16* dstK /*layer0: A1r (K half)*/, __bf16* dstH /*next-step hidden half*/,
    float* h1f, float (*Gbuf)[32][32]) {
  const int t = threadIdx.x;
  const int wave = t >> 5, lane = t & 31;
  const int gi = wave >> 1, jt = wave & 1;     // gate, j-tile
  const int m16 = lane & 15, hi = lane >> 4;
  const int nt = gi * 64 + (int)blockIdx.x * 2 + jt;

  const __bf16* bp  = Wb + ((size_t)l * NT + nt) * KT * 512 + lane * 16;
  const __bf16* a0p = Ag + (size_t)m16 * K2 + hi * 16;
  const __bf16* a1p = Ag + (size_t)(16 + m16) * K2 + hi * 16;

  f32x8 acc0 = {}, acc1 = {};
  for (int kt = 0; kt < KT; ++kt) {
    V16 bv, a0v, a1v;
    bv.q[0]  = *(const u32x4*)(bp);          // B frag: 32B contiguous per lane
    bv.q[1]  = *(const u32x4*)(bp + 8);
    a0v.q[0] = *(const u32x4*)(a0p);         // A frag mt=0 (contiguous k = 16*hi + e, matches B)
    a0v.q[1] = *(const u32x4*)(a0p + 8);
    a1v.q[0] = *(const u32x4*)(a1p);         // A frag mt=1
    a1v.q[1] = *(const u32x4*)(a1p + 8);
    acc0 = wmma_bf16(a0v.v, bv.v, acc0);
    acc1 = wmma_bf16(a1v.v, bv.v, acc1);
    bp += 512; a0p += 32; a1p += 32;
  }

  // D layout: lane n = L&15, VGPR v -> row m = v + (L>=16 ? 8 : 0)
  const int jl = jt * 16 + m16;
#pragma unroll
  for (int v = 0; v < 8; ++v) {
    Gbuf[gi][v + hi * 8][jl]      = acc0[v];
    Gbuf[gi][16 + v + hi * 8][jl] = acc1[v];
  }
  __syncthreads();

  const float* bi = b_ih + (size_t)l * NG;
  const float* bh = b_hh + (size_t)l * NG;
  for (int it = 0; it < 4; ++it) {
    int idx = t + it * TPB;
    int m = idx >> 5, j = idx & 31;
    int col = (int)blockIdx.x * 32 + j;
    float xi = Gbuf[0][m][j] + bi[col]          + bh[col];
    float xf = Gbuf[1][m][j] + bi[H_ + col]     + bh[H_ + col];
    float xg = Gbuf[2][m][j] + bi[2 * H_ + col] + bh[2 * H_ + col];
    float xo = Gbuf[3][m][j] + bi[3 * H_ + col] + bh[3 * H_ + col];
    float* cp = cst + ((size_t)l * B_ + m) * H_ + col;
    float si = 1.f / (1.f + __expf(-xi));
    float sf = 1.f / (1.f + __expf(-xf));
    float so = 1.f / (1.f + __expf(-xo));
    float cn = sf * cp[0] + si * tanhf(xg);
    float hn = so * tanhf(cn);
    cp[0] = cn;
    __bf16 hb = (__bf16)hn;
    if (l == 0) {
      dstK[m * K2 + col]      = hb;   // K input for layer 1 this step
      dstH[m * K2 + H_ + col] = hb;   // hidden half for next step's layer 0
    } else {
      dstH[m * K2 + H_ + col] = hb;   // hidden half for next step's layer 1
      h1f[m * H_ + col]       = hn;   // fp32 copy for the softmax head
    }
  }
}

// y[b,:] = softmax(h1[b,:] @ w_post^T + b_post); write ws.y and d_out[b,s,:]
__device__ __forceinline__ void phase_y(int b, int s, int dec, const float* h1f,
                                        const float* w_post, const float* b_post,
                                        float* yg, float* out, float* sm) {
  const int t = threadIdx.x;
  const int c = t & 63, part = t >> 6;
  float p = 0.f;
  if (c < NC_) {
    const float* hr = h1f + (size_t)b * H_ + part * 256;
    const float* wr = w_post + (size_t)c * H_ + part * 256;
    for (int k = 0; k < 256; ++k) p = fmaf(hr[k], wr[k], p);
  }
  sm[c * 4 + part] = p;
  __syncthreads();
  float* Ls = sm + 256;
  if (part == 0)
    Ls[c] = (c < NC_) ? (sm[c * 4] + sm[c * 4 + 1] + sm[c * 4 + 2] + sm[c * 4 + 3]
                         + b_post[c])
                      : -1e30f;
  __syncthreads();
  float* red2 = sm + 320;
  if (t == 0) {
    float mx = -1e30f;
    for (int i = 0; i < NC_; ++i) mx = fmaxf(mx, Ls[i]);
    float ssum = 0.f;
    for (int i = 0; i < NC_; ++i) ssum += __expf(Ls[i] - mx);
    red2[0] = mx; red2[1] = 1.f / ssum;
  }
  __syncthreads();
  if (t < NC_) {
    float yv = __expf(Ls[t] - red2[0]) * red2[1];
    yg[b * 64 + t] = yv;
    out[((size_t)b * dec + s) * NC_ + t] = yv;
  }
  __syncthreads();
}

__global__ __launch_bounds__(TPB) void
lstm_decoder_kernel(const float* __restrict__ hs, const float* __restrict__ w_ch2,
                    const float* __restrict__ b_ch2, const float* __restrict__ w_ch1,
                    const float* __restrict__ b_ch1, const float* __restrict__ w_cc1,
                    const float* __restrict__ b_cc1, const float* __restrict__ w_ih,
                    const float* __restrict__ w_hh, const float* __restrict__ b_ih,
                    const float* __restrict__ b_hh, const float* __restrict__ w_pre,
                    const float* __restrict__ b_pre, const float* __restrict__ w_post,
                    const float* __restrict__ b_post, const int* __restrict__ decp,
                    float* __restrict__ out, char* __restrict__ ws) {
  unsigned* bar = (unsigned*)(ws + BAR_OFF);
  float* yg  = (float*)(ws + Y_OFF);
  float* h1f = (float*)(ws + H1F_OFF);
  float* cst = (float*)(ws + C_OFF);
  float* U   = (float*)(ws + U_OFF);
  float* Sg  = (float*)(ws + S_OFF);
  __bf16* A0 = (__bf16*)(ws + A0_OFF);
  __bf16* A1 = (__bf16*)(ws + A1_OFF);
  __bf16* Wb = (__bf16*)(ws + WB_OFF);

  __shared__ float Gbuf[4][32][32];   // 16 KB gate exchange / reduction scratch

  const int b = blockIdx.x;           // WG <-> batch row for scalar phases
  const int t = threadIdx.x;
  int dec = *decp;
  if (dec < 1) dec = 1;
  if (dec > 1300) dec = 1300;         // barrier-slot capacity guard
  int bidx = 0;

  // ---- prologue 1: u[l,b,:] = sum_t w[l,t] * hs[b,t,:] (hs read once) ----
  for (int i = 0; i < 4; ++i) {
    int h = t + i * TPB;
    const float* hp = hs + (size_t)b * T_ * H_ + h;
    float uh0 = 0, uh1 = 0, uc0 = 0, uc1 = 0;
    for (int tt = 0; tt < T_; ++tt) {
      float x = hp[(size_t)tt * H_];
      uh0 = fmaf(x, w_ch1[tt],      uh0);
      uh1 = fmaf(x, w_ch1[T_ + tt], uh1);
      uc0 = fmaf(x, w_cc1[tt],      uc0);
      uc1 = fmaf(x, w_cc1[T_ + tt], uc1);
    }
    U[((0 * NL_ + 0) * B_ + b) * H_ + h] = uh0;
    U[((0 * NL_ + 1) * B_ + b) * H_ + h] = uh1;
    U[((1 * NL_ + 0) * B_ + b) * H_ + h] = uc0;
    U[((1 * NL_ + 1) * B_ + b) * H_ + h] = uc1;
  }
  if (b == 0 && t < 4) {              // S[l] = sum_t w_ch1[l,t] (and w_cc1)
    const float* src = (t < 2) ? (w_ch1 + t * T_) : (w_cc1 + (t - 2) * T_);
    float s = 0;
    for (int i = 0; i < T_; ++i) s += src[i];
    Sg[t] = s;
  }
  if (t < NC_) {                      // y0 = one_hot(0)
    float v = (t == 0) ? 1.f : 0.f;
    yg[b * 64 + t] = v;
    out[((size_t)b * dec + 0) * NC_ + t] = v;
  }
  // ---- prologue 1b: swizzle fp32 weights -> bf16 WMMA-B fragment layout ----
  // block(l,nt,kt): lane L holds k = kt*32 + (L>>4)*16 + e, n = nt*16 + (L&15)
  {
    const size_t NPAIR = (size_t)NL_ * NT * KT * 32 * 8;   // 2-element packs
    size_t gid = (size_t)b * TPB + t;
    for (size_t p = gid; p < NPAIR; p += (size_t)NWG * TPB) {
      int e    = (int)(p & 7) * 2;
      int lane = (int)((p >> 3) & 31);
      int kt   = (int)((p >> 8) & 63);
      int nt   = (int)((p >> 14) & 255);
      int l    = (int)(p >> 22);
      int n = nt * 16 + (lane & 15);
      int k = kt * 32 + (lane >> 4) * 16 + e;
      const float* srcw = (k < H_) ? (w_ih + ((size_t)l * NG + n) * H_ + k)
                                   : (w_hh + ((size_t)l * NG + n) * H_ + (k - H_));
      union { __bf16 h2[2]; unsigned u; } pk;
      pk.h2[0] = (__bf16)srcw[0];
      pk.h2[1] = (__bf16)srcw[1];
      *(unsigned*)(Wb + ((((size_t)l * NT + nt) * KT + kt) * 32 + lane) * 16 + e) = pk.u;
    }
  }
  grid_sync(bar, bidx++);

  // ---- prologue 2: h0/c0 via folded context GEMM; init step-1 A buffers ----
  {
    __bf16* A0w = A0 + (size_t)1 * B_ * K2;   // step 1 uses buffer parity 1
    __bf16* A1w = A1 + (size_t)1 * B_ * K2;
    float Sh0 = Sg[0], Sh1 = Sg[1], Sc0 = Sg[2], Sc1 = Sg[3];
    const float* uh0p = U + ((0 * NL_ + 0) * B_ + b) * H_;
    const float* uh1p = U + ((0 * NL_ + 1) * B_ + b) * H_;
    const float* uc0p = U + ((1 * NL_ + 0) * B_ + b) * H_;
    const float* uc1p = U + ((1 * NL_ + 1) * B_ + b) * H_;
    for (int i = 0; i < 4; ++i) {
      int h = t + i * TPB;
      const float* wr = w_ch2 + (size_t)h * H_;
      float ah0 = 0, ah1 = 0, ac0 = 0, ac1 = 0;
      for (int k = 0; k < H_; ++k) {
        float w = wr[k];
        ah0 = fmaf(w, uh0p[k], ah0);
        ah1 = fmaf(w, uh1p[k], ah1);
        ac0 = fmaf(w, uc0p[k], ac0);
        ac1 = fmaf(w, uc1p[k], ac1);
      }
      float bb = b_ch2[h];
      float h0v = ah0 + bb * Sh0 + b_ch1[0];
      float h1v = ah1 + bb * Sh1 + b_ch1[1];
      float c0v = ac0 + bb * Sc0 + b_cc1[0];
      float c1v = ac1 + bb * Sc1 + b_cc1[1];
      A0w[b * K2 + H_ + h] = (__bf16)h0v;
      A1w[b * K2 + H_ + h] = (__bf16)h1v;
      h1f[b * H_ + h] = h1v;
      cst[(0 * B_ + b) * H_ + h] = c0v;
      cst[(1 * B_ + b) * H_ + h] = c1v;
    }
    compute_inp(b, yg, w_pre, b_pre, A0w);    // inp for step 1 from y0
  }
  grid_sync(bar, bidx++);

  // ---- the scan: 3 grid barriers per step ----
  for (int s = 1; s < dec; ++s) {
    int p = s & 1;
    __bf16* A0r = A0 + (size_t)p * B_ * K2;
    __bf16* A0w = A0 + (size_t)(1 - p) * B_ * K2;
    __bf16* A1r = A1 + (size_t)p * B_ * K2;
    __bf16* A1w = A1 + (size_t)(1 - p) * B_ * K2;

    lstm_layer(0, A0r, Wb, b_ih, b_hh, cst, A1r, A0w, h1f, Gbuf);
    grid_sync(bar, bidx++);
    lstm_layer(1, A1r, Wb, b_ih, b_hh, cst, A1r, A1w, h1f, Gbuf);
    grid_sync(bar, bidx++);
    phase_y(b, s, dec, h1f, w_post, b_post, yg, out, &Gbuf[0][0][0]);
    compute_inp(b, yg, w_pre, b_pre, A0w);    // inp for step s+1
    grid_sync(bar, bidx++);
  }
}

extern "C" void kernel_launch(void* const* d_in, const int* in_sizes, int n_in,
                              void* d_out, int out_size, void* d_ws, size_t ws_size,
                              hipStream_t stream) {
  (void)in_sizes; (void)n_in; (void)out_size; (void)ws_size;
  // zero the one-shot barrier counters (graph-capture-safe memset node)
  hipMemsetAsync(d_ws, 0, 16384, stream);
  lstm_decoder_kernel<<<dim3(NWG), dim3(TPB), 0, stream>>>(
      (const float*)d_in[0],  (const float*)d_in[1],  (const float*)d_in[2],
      (const float*)d_in[3],  (const float*)d_in[4],  (const float*)d_in[5],
      (const float*)d_in[6],  (const float*)d_in[7],  (const float*)d_in[8],
      (const float*)d_in[9],  (const float*)d_in[10], (const float*)d_in[11],
      (const float*)d_in[12], (const float*)d_in[13], (const float*)d_in[14],
      (const int*)d_in[15],   (float*)d_out,          (char*)d_ws);
}
